// MVAdapt_65884798321210
// MI455X (gfx1250) — matrix operand order repaired and checked by reading, not compiled, not run on hardware
//
#include <hip/hip_runtime.h>
#include <hip/hip_bf16.h>

// ---------------------------------------------------------------------------
// Hypernetwork MLP, fused, reordered contraction + 2-sample-group A reuse:
//   y[b,o] = sum_k h[b,k] * t[o,k,b],   t[o,k,b] = sum_i E[(o,i),k] x[b,i]
// Per block: 32 samples = 2 WMMA sample groups. Every e_w2 A operand loaded
// once feeds 2 WMMAs (one per group) -> halves L2 traffic vs 16-sample blocks.
// Stage 1 (per output unit o): 4 C tiles x (D_in/32) ksteps per group of
// v_wmma_f32_16x16x32_bf16, processed as 2 mt-passes to bound VGPRs.
// Stage 2: 32 FMAs vs h (f32, registers) + shfl_xor + one ds_add_f32 per o.
// e_b2-weight term and generated biases = tiny extra WMMA tiles.
// e_w2 pre-swizzled to bf16 A-operand layout (~4.7MB, L2 resident).
// ---------------------------------------------------------------------------

typedef __attribute__((ext_vector_type(16))) __bf16 bf16x16;
typedef __attribute__((ext_vector_type(8)))  float  floatx8;

#define BATCH    2048
#define SPB      16        // samples per WMMA group (N)
#define SG       2         // sample groups per block
#define NWAVES   8

// per-layer constants
// DIN  = {16,64,128,128,64}   DOUT = {64,128,128,64,16}
// KS   = ceil(DIN/32) = {1,2,4,4,2}
// WOFF = {0,1088,9408,25920,34176}   BOFF = {1024,9280,25792,34112,35200}
// operand-segment bases (units of 512 bf16 = one A operand for 32 lanes):
//   layer: [main: DOUT*4*KS | eb2: (DOUT/16)*KS | bias: (DOUT/16)*2]
#define NOPS_TOTAL 4600

__device__ __constant__ int c_lbase[6] = {0, 268, 1324, 3420, 4468, 4600};
__device__ __constant__ int c_din [5] = {16, 64, 128, 128, 64};
__device__ __constant__ int c_dout[5] = {64, 128, 128, 64, 16};
__device__ __constant__ int c_ks  [5] = {1, 2, 4, 4, 2};
__device__ __constant__ int c_woff[5] = {0, 1088, 9408, 25920, 34176};
__device__ __constant__ int c_boff[5] = {1024, 9280, 25792, 34112, 35200};

static __device__ __forceinline__ floatx8 wmma_bf16(bf16x16 a, bf16x16 b, floatx8 c) {
    return __builtin_amdgcn_wmma_f32_16x16x32_bf16(false, a, false, b,
                                                   (short)0, c, false, false);
}

// --------------------------- prep: build swizzled bf16 A operands ----------
// A-operand layout (16x32 bf16, ISA 7.12.2): lane m(0-15): row m, cols
// {0..7,16..23}; lane m+16: row m, cols {8..15,24..31}.
__global__ __launch_bounds__(256)
void prep_kernel(const float* __restrict__ e_w2, const float* __restrict__ e_b2,
                 __bf16* __restrict__ abuf) {
    int t = blockIdx.x * blockDim.x + threadIdx.x;
    if (t >= NOPS_TOTAL * 32) return;
    int lane = t & 31;
    int op   = t >> 5;
    int l = 0;
    while (l < 4 && op >= c_lbase[l + 1]) ++l;
    int q = op - c_lbase[l];
    const int DIN = c_din[l], DOUT = c_dout[l], KS = c_ks[l];
    const int WOFF = c_woff[l], BOFF = c_boff[l];
    const int MAIN = DOUT * 4 * KS;
    const int EBN  = (DOUT / 16) * KS;
    const int m  = lane & 15;
    const int kh = lane >> 4;
    __bf16* dst = abuf + (size_t)op * 512 + lane * 16;

    if (q < MAIN) {
        // transposed main operand: row = k, col = i  (A[m][i] = E[(o,i),k])
        int o   = q / (4 * KS);
        int rem = q - o * 4 * KS;
        int mt  = rem / KS;
        int ks  = rem - mt * KS;
        int k   = mt * 16 + m;
        #pragma unroll
        for (int j = 0; j < 16; ++j) {
            int il = ((j < 8) ? j : (j + 8)) + kh * 8;
            int i  = ks * 32 + il;
            float v = (i < DIN) ? e_w2[(size_t)(WOFF + o * DIN + i) * 64 + k] : 0.0f;
            dst[j] = (__bf16)v;
        }
    } else if (q < MAIN + EBN) {
        // eb2-weight operand: row = o (16 of them), col = i
        int q2 = q - MAIN;
        int ot = q2 / KS;
        int ks = q2 - ot * KS;
        int o  = ot * 16 + m;
        #pragma unroll
        for (int j = 0; j < 16; ++j) {
            int il = ((j < 8) ? j : (j + 8)) + kh * 8;
            int i  = ks * 32 + il;
            float v = (i < DIN) ? e_b2[WOFF + o * DIN + i] : 0.0f;
            dst[j] = (__bf16)v;
        }
    } else {
        // bias operand: row = bias index, col = k (original orientation)
        int q3 = q - MAIN - EBN;
        int bt = q3 >> 1;
        int s  = q3 & 1;
        int row = BOFF + bt * 16 + m;
        #pragma unroll
        for (int j = 0; j < 16; ++j) {
            int kl = ((j < 8) ? j : (j + 8)) + kh * 8;
            dst[j] = (__bf16)e_w2[(size_t)row * 64 + s * 32 + kl];
        }
    }
}

// --------------------------- fused layer ----------------------------------
template <int DIN, int DOUT, int KS, int MB, int EBB, int BIB, int BOFF>
__device__ __forceinline__ void run_layer(
    const __bf16* __restrict__ abuf, const float* __restrict__ eb2,
    __bf16 (*Xbf)[SPB][128], float (*y_lds)[128][SPB], float (*Hf)[SPB][64],
    __bf16 (*Hbf)[SPB][64], int tid, int lane, int wid)
{
    for (int e = tid; e < SG * DOUT * SPB; e += 256) ((float*)y_lds)[e] = 0.0f;

    // B operands (x, bf16) per group, registers for whole layer:
    // lane<16: col n=lane, i = ks*32 + {0..7,16..23}; lane>=16: i = ks*32 + {8..15,24..31}.
    const int n    = lane & 15;
    const int kh16 = (lane >> 4) * 16;
    bf16x16 XB[SG][KS];
    #pragma unroll
    for (int g = 0; g < SG; ++g)
        #pragma unroll
        for (int ks = 0; ks < KS; ++ks)
            #pragma unroll
            for (int j = 0; j < 16; ++j)
                XB[g][ks][j] = Xbf[g][n][ks * 32 + (lane >> 4) * 8 + ((j < 8) ? j : (j + 8))];

    // h in f32 registers, matching the t C-tile layout:
    // hreg[g][mt*8+r] = h[b = lane&15][k = mt*16 + r + (lane>>4)*8]
    float hreg[SG][32];
    #pragma unroll
    for (int g = 0; g < SG; ++g)
        #pragma unroll
        for (int j = 0; j < 32; ++j)
            hreg[g][j] = Hf[g][n][(j >> 3) * 16 + (j & 7) + (lane >> 4) * 8];

    __syncthreads();

    const int rlane = (lane >> 4) * 8;
    const bf16x16* A = (const bf16x16*)abuf;
    constexpr int NE = DOUT / 16;
    constexpr int NWORK = DOUT + NE + NE;

    for (int T = wid; T < NWORK; T += NWAVES) {
        if (T < DOUT) {
            // ---- main: t[o,:,:] = E_o^T (64 x DIN) @ x (DIN x 16), x2 grp --
            const int o    = T;
            const int base = MB + o * 4 * KS;
            float sum[SG] = {0.0f, 0.0f};
            #pragma unroll
            for (int mtp = 0; mtp < 2; ++mtp) {     // mt pair: {0,1} / {2,3}
                floatx8 c[SG][2] = {};
                #pragma unroll
                for (int ks = 0; ks < KS; ++ks) {
                    bf16x16 a0 = A[(size_t)(base + (2 * mtp + 0) * KS + ks) * 32 + lane];
                    bf16x16 a1 = A[(size_t)(base + (2 * mtp + 1) * KS + ks) * 32 + lane];
                    c[0][0] = wmma_bf16(a0, XB[0][ks], c[0][0]);
                    c[1][0] = wmma_bf16(a0, XB[1][ks], c[1][0]);
                    c[0][1] = wmma_bf16(a1, XB[0][ks], c[0][1]);
                    c[1][1] = wmma_bf16(a1, XB[1][ks], c[1][1]);
                }
                #pragma unroll
                for (int g = 0; g < SG; ++g) {
                    float s = 0.0f;
                    #pragma unroll
                    for (int mth = 0; mth < 2; ++mth)
                        #pragma unroll
                        for (int r = 0; r < 8; ++r)
                            s += hreg[g][(2 * mtp + mth) * 8 + r] * c[g][mth][r];
                    sum[g] += s;
                }
            }
            #pragma unroll
            for (int g = 0; g < SG; ++g) {
                float s = sum[g] + __shfl_xor(sum[g], 16);   // other half of k
                if (lane < 16) atomicAdd(&y_lds[g][o][lane], s);
            }
        } else if (T < DOUT + NE) {
            // ---- eb2-weight term: sum_i e_b2[(o,i)] x[b,i], 16 o's x2 grp --
            const int ot   = T - DOUT;
            const int base = EBB + ot * KS;
            floatx8 c[SG] = {};
            #pragma unroll
            for (int ks = 0; ks < KS; ++ks) {
                bf16x16 a = A[(size_t)(base + ks) * 32 + lane];
                c[0] = wmma_bf16(a, XB[0][ks], c[0]);
                c[1] = wmma_bf16(a, XB[1][ks], c[1]);
            }
            #pragma unroll
            for (int g = 0; g < SG; ++g)
                #pragma unroll
                for (int r = 0; r < 8; ++r)
                    atomicAdd(&y_lds[g][ot * 16 + rlane + r][n], c[g][r]);
        } else {
            // ---- generated biases: h @ E_bias^T + e_b2[bias], x2 groups ---
            const int bt   = T - DOUT - NE;
            const int base = BIB + bt * 2;
            bf16x16 a0 = A[(size_t)(base + 0) * 32 + lane];
            bf16x16 a1 = A[(size_t)(base + 1) * 32 + lane];
            const float* eb = eb2 + BOFF + bt * 16 + rlane;
            float4 e0 = *(const float4*)eb;
            float4 e1 = *(const float4*)(eb + 4);
            float ebv[8] = {e0.x, e0.y, e0.z, e0.w, e1.x, e1.y, e1.z, e1.w};
            #pragma unroll
            for (int g = 0; g < SG; ++g) {
                bf16x16 hb0, hb1;                  // rebuild from LDS (rare)
                #pragma unroll
                for (int j = 0; j < 16; ++j) {
                    hb0[j] = Hbf[g][n][(lane >> 4) * 8 + ((j < 8) ? j : (j + 8))];
                    hb1[j] = Hbf[g][n][32 + (lane >> 4) * 8 + ((j < 8) ? j : (j + 8))];
                }
                floatx8 c = {};
                c = wmma_bf16(a0, hb0, c);
                c = wmma_bf16(a1, hb1, c);
                #pragma unroll
                for (int r = 0; r < 8; ++r)
                    atomicAdd(&y_lds[g][bt * 16 + rlane + r][n], c[r] + ebv[r]);
            }
        }
    }
    __syncthreads();
}

// --------------------------- main fused kernel -----------------------------
__global__ __launch_bounds__(256)
void hyper_kernel(const float* __restrict__ wp,   const float* __restrict__ phys,
                  const float* __restrict__ gear, const float* __restrict__ g_w1,
                  const float* __restrict__ g_b1, const float* __restrict__ g_w2,
                  const float* __restrict__ g_b2, const float* __restrict__ e_w1,
                  const float* __restrict__ e_b1, const float* __restrict__ eb2,
                  const __bf16* __restrict__ abuf, float* __restrict__ out)
{
    __shared__ __bf16 Xbf[SG][SPB][128];         // activations, bf16 B source
    __shared__ float  y_lds[SG][128][SPB];
    __shared__ float  Hf[SG][SPB][64];           // h, f32 (stage-2 precision)
    __shared__ __bf16 Hbf[SG][SPB][64];          // h, bf16 (bias-tile B src)
    __shared__ float  ph_lds[SG * SPB][22];      // physics(18) ++ gear_out(4)
    __shared__ float  gear_lds[SG * SPB][24];
    __shared__ float  g1_lds[SG * SPB][8];

    const int tid  = threadIdx.x;
    const int lane = tid & 31;
    const int wid  = tid >> 5;
    const int b0g  = blockIdx.x * (SG * SPB);

    // ---- stage inputs (32 samples) ----
    for (int e = tid; e < SG * SPB * 24; e += 256) {
        int s = e / 24, j = e % 24;
        gear_lds[s][j] = gear[(size_t)(b0g + s) * 24 + j];
    }
    for (int e = tid; e < SG * SPB * 18; e += 256) {
        int s = e / 18, j = e % 18;
        ph_lds[s][j] = phys[(size_t)(b0g + s) * 18 + j];
    }
    for (int e = tid; e < SG * SPB * 32; e += 256) {  // x0 = waypoint, pad 16..31
        int s = e >> 5, j = e & 31;
        Xbf[s >> 4][s & 15][j] =
            (j < 16) ? (__bf16)wp[(size_t)(b0g + s) * 16 + j] : (__bf16)0.0f;
    }
    __syncthreads();

    // ---- gear MLP: 24 -> 8 -> 4 (tiny, VALU) ----
    {   // 32 samples x 8 units = 256 outputs
        int s = tid >> 3, u = tid & 7;
        float a = g_b1[u];
        #pragma unroll
        for (int k = 0; k < 24; ++k) a += gear_lds[s][k] * g_w1[u * 24 + k];
        g1_lds[s][u] = fmaxf(a, 0.0f);
    }
    __syncthreads();
    if (tid < 128) {                             // 32 samples x 4 units
        int s = tid >> 2, u = tid & 3;
        float a = g_b2[u];
        #pragma unroll
        for (int k = 0; k < 8; ++k) a += g1_lds[s][k] * g_w2[u * 8 + k];
        ph_lds[s][18 + u] = fmaxf(a, 0.0f);
    }
    __syncthreads();

    // ---- encoder hidden h: 22 -> 64, relu; keep f32 + bf16 copies ----
    #pragma unroll
    for (int rep = 0; rep < 8; ++rep) {          // 32 samples x 64 units
        int idx = tid + rep * 256;
        int s = idx >> 6, u = idx & 63;
        float a = e_b1[u];
        #pragma unroll
        for (int k = 0; k < 22; ++k) a += ph_lds[s][k] * e_w1[u * 22 + k];
        a = fmaxf(a, 0.0f);
        Hf[s >> 4][s & 15][u]  = a;
        Hbf[s >> 4][s & 15][u] = (__bf16)a;
    }
    __syncthreads();

    // ---- fused hyper-layers ----
    run_layer< 16,  64, 1,    0,  256,  260,  1024>(abuf, eb2, Xbf, y_lds, Hf, Hbf, tid, lane, wid);
    for (int e = tid; e < SG * 64 * SPB; e += 256) {
        int g = e >> 10, o = (e >> 4) & 63, b = e & 15;
        Xbf[g][b][o] = (__bf16)fmaxf(y_lds[g][o][b], 0.0f);
    }
    __syncthreads();

    run_layer< 64, 128, 2,  268, 1292, 1308,  9280>(abuf, eb2, Xbf, y_lds, Hf, Hbf, tid, lane, wid);
    for (int e = tid; e < SG * 128 * SPB; e += 256) {
        int g = e >> 11, o = (e >> 4) & 127, b = e & 15;
        Xbf[g][b][o] = (__bf16)fmaxf(y_lds[g][o][b], 0.0f);
    }
    __syncthreads();

    run_layer<128, 128, 4, 1324, 3372, 3404, 25792>(abuf, eb2, Xbf, y_lds, Hf, Hbf, tid, lane, wid);
    for (int e = tid; e < SG * 128 * SPB; e += 256) {
        int g = e >> 11, o = (e >> 4) & 127, b = e & 15;
        Xbf[g][b][o] = (__bf16)fmaxf(y_lds[g][o][b], 0.0f);
    }
    __syncthreads();

    run_layer<128,  64, 4, 3420, 4444, 4460, 34112>(abuf, eb2, Xbf, y_lds, Hf, Hbf, tid, lane, wid);
    for (int e = tid; e < SG * 64 * SPB; e += 256) {
        int g = e >> 10, o = (e >> 4) & 63, b = e & 15;
        Xbf[g][b][o] = (__bf16)fmaxf(y_lds[g][o][b], 0.0f);
    }
    __syncthreads();

    run_layer< 64,  16, 2, 4468, 4596, 4598, 35200>(abuf, eb2, Xbf, y_lds, Hf, Hbf, tid, lane, wid);

    // ---- final output (no relu): out[b][o] ----
    for (int e = tid; e < SG * 16 * SPB; e += 256) {
        int g = e >> 8, o = (e >> 4) & 15, b = e & 15;
        out[(size_t)(b0g + g * SPB + b) * 16 + o] = y_lds[g][o][b];
    }
}

// --------------------------- launch ----------------------------------------
extern "C" void kernel_launch(void* const* d_in, const int* in_sizes, int n_in,
                              void* d_out, int out_size, void* d_ws, size_t ws_size,
                              hipStream_t stream) {
    (void)in_sizes; (void)n_in; (void)out_size; (void)ws_size;
    const float* wp   = (const float*)d_in[0];
    const float* phys = (const float*)d_in[1];
    const float* gear = (const float*)d_in[2];
    const float* g_w1 = (const float*)d_in[3];
    const float* g_b1 = (const float*)d_in[4];
    const float* g_w2 = (const float*)d_in[5];
    const float* g_b2 = (const float*)d_in[6];
    const float* e_w1 = (const float*)d_in[7];
    const float* e_b1 = (const float*)d_in[8];
    const float* e_w2 = (const float*)d_in[9];
    const float* e_b2 = (const float*)d_in[10];

    __bf16* abuf = (__bf16*)d_ws;                // 4600 * 512 bf16 = 4.71 MB

    const int nprep = NOPS_TOTAL * 32;
    prep_kernel<<<(nprep + 255) / 256, 256, 0, stream>>>(e_w2, e_b2, abuf);

    hyper_kernel<<<BATCH / (SG * SPB), 256, 0, stream>>>(
        wp, phys, gear, g_w1, g_b1, g_w2, g_b2, e_w1, e_b1, e_b2, abuf,
        (float*)d_out);
}
